// CrossStructureModule_84172769067167
// MI455X (gfx1250) — compile-verified, matrix operand order checked by reading, not run on hardware
//
#include <hip/hip_runtime.h>
#include <math.h>

// =====================================================================
// CrossStructureModule forward for MI455X (gfx1250, wave32, WMMA).
// GEMMs: f16-in/f32-accum v_wmma_f32_16x16x32_f16, K templated so the
// K loop fully unrolls (many static WMMAs, software-pipelined loads).
// Data path per K-step: coalesced b128 global loads -> LDS (B stored
// transposed so WMMA fragments are contiguous) -> ds_load_b128 -> WMMA.
// The net is HBM/latency bound (~15 GFLOP vs ~150MB traffic); f16
// operands are safe (LayerNorm-bounded activations) and give 8x the
// K-depth of the f32 16x16x4 WMMA. Loop-invariant protein projections
// (kv, kvp) are hoisted out of the NBLOCKS loop.
// =====================================================================

typedef __attribute__((ext_vector_type(16))) _Float16 v16h;
typedef __attribute__((ext_vector_type(8)))  _Float16 v8h;
typedef __attribute__((ext_vector_type(8)))  float    v8f;

#define WAVE 32

// ---- model constants ------------------------------------------------
#define Bb    64
#define LQ    16
#define LP    200
#define CS    384
#define CIPA  16
#define CRES  128
#define Hh    12
#define PQ    4
#define PV    8
#define NANG  7
#define NGRP  8
#define NAT   18
#define NTOK  (Bb*LQ)      // 1024
#define NPTOK (Bb*LP)      // 12800
#define FEATD (Hh*(CIPA + PV*4))  // 576

// ---- input index map (jax tree_flatten: sorted dict keys) -----------
enum {
  IN_T_PROTEIN = 0,
  IN_ATOM_MASK, IN_DEF_FRAMES, IN_GROUP_IDX, IN_LIT_POS,   // buffers (sorted)
  IN_AATYPE, IN_LOOP_MASK,
  IN_ANG_B0_L1_B, IN_ANG_B0_L1_W, IN_ANG_B0_L2_B, IN_ANG_B0_L2_W,
  IN_ANG_B1_L1_B, IN_ANG_B1_L1_W, IN_ANG_B1_L2_B, IN_ANG_B1_L2_W,
  IN_ANG_IN_B, IN_ANG_IN_W, IN_ANG_INIT_B, IN_ANG_INIT_W,
  IN_ANG_OUT_B, IN_ANG_OUT_W,
  IN_BB_B, IN_BB_W,
  IN_INLOOP_B, IN_INLOOP_W, IN_INPROT_B, IN_INPROT_W,
  IN_HEAD_W, IN_KV_B, IN_KV_W, IN_KVP_B, IN_KVP_W,
  IN_IPAOUT_B, IN_IPAOUT_W, IN_Q_B, IN_Q_W, IN_QP_B, IN_QP_W,
  IN_LNIPA_B, IN_LNIPA_G, IN_LNLOOP_B, IN_LNLOOP_G,
  IN_LNPROT_B, IN_LNPROT_G, IN_LNTRANS_B, IN_LNTRANS_G,
  IN_TL_L1_B, IN_TL_L1_W, IN_TL_L2_B, IN_TL_L2_W, IN_TL_L3_B, IN_TL_L3_W,
  IN_PROT_MASK, IN_S_LOOP_INIT, IN_S_PROT_INIT
};

// ---- wave helpers ---------------------------------------------------
__device__ __forceinline__ float wave_sum(float v){
#pragma unroll
  for (int o = 16; o > 0; o >>= 1) v += __shfl_xor(v, o, WAVE);
  return v;
}
__device__ __forceinline__ float wave_max(float v){
#pragma unroll
  for (int o = 16; o > 0; o >>= 1) v = fmaxf(v, __shfl_xor(v, o, WAVE));
  return v;
}
__device__ __forceinline__ void quat_rot(const float* __restrict__ q,
                                         float* __restrict__ R){
  float w=q[0], x=q[1], y=q[2], z=q[3];
  R[0]=1.f-2.f*(y*y+z*z); R[1]=2.f*(x*y-w*z);      R[2]=2.f*(x*z+w*y);
  R[3]=2.f*(x*y+w*z);     R[4]=1.f-2.f*(x*x+z*z);  R[5]=2.f*(y*z-w*x);
  R[6]=2.f*(x*z-w*y);     R[7]=2.f*(y*z+w*x);      R[8]=1.f-2.f*(x*x+y*y);
}

// =====================================================================
// Tiled WMMA GEMM: Y[M,N] = act(X[M,K]) @ W[K,N] + bias[N] (+res) (relu?)
// Block = 128 threads = 4 waves sharing one 16-row A tile (blockIdx.y)
// and covering 64 columns (blockIdx.x). K is a template constant so the
// main loop fully unrolls into K/32 v_wmma_f32_16x16x32_f16 ops.
// LDS staging:
//   As[16][32]  f16, row-major (A tile, cooperatively loaded as float4)
//   Bs[64][32]  f16, W tile stored TRANSPOSED (n-major) so each wave's
//               fragment is two contiguous 8-half runs -> ds_load_b128.
// Fragment packing per CDNA5 ISA wave32 layouts:
//   A: row=lane%16, K = kb+{0..7} and kb+16+{0..7}, kb=8*(lane/16)
//   C: element r -> row = 8*(lane/16)+r, col = lane%16
// Requires M%16==0, K%32==0; partial N handled with clamped loads and
// wave-uniform tile guards (EXEC stays all-ones around every WMMA).
// =====================================================================
template<int K, bool IN_RELU, bool OUT_RELU>
__global__ void gemm_tile_k(const float* __restrict__ X,
                            const float* __restrict__ W,
                            const float* __restrict__ bias,
                            const float* __restrict__ res,
                            float* __restrict__ Y, int M, int N)
{
  __shared__ __align__(16) _Float16 As[16 * 32];
  __shared__ __align__(16) _Float16 Bs[64 * 32];

  const int tid   = threadIdx.x;
  const int lane  = tid & 31;
  const int wv    = tid >> 5;
  const int l16   = lane & 15;
  const int half  = lane >> 4;
  const int mtile = blockIdx.y;
  const int nbase = blockIdx.x * 64;
  const int ntile = nbase + wv * 16;
  const bool active = (ntile < N);             // wave-uniform
  const bool fullN  = (nbase + 64 <= N);       // block-uniform

  // A-tile cooperative load coords: 16 rows x 8 float4 segments
  const int arow = tid >> 3;          // 0..15
  const int aseg = (tid & 7) * 4;     // 0,4,..,28
  const float* __restrict__ xr = X + ((size_t)(mtile * 16 + arow)) * K;
  // B-tile cooperative load coords: per pass 8 k-rows x 16 float4 segs
  const int bkr = tid >> 4;           // 0..7
  const int bns = (tid & 15) * 4;     // 0,4,..,60

  v8f acc = {};
#pragma unroll
  for (int k0 = 0; k0 < K; k0 += 32) {
    // ---- stage A (16x32 f32 -> f16) ----
    {
      float4 a4 = *(const float4*)(xr + k0 + aseg);
      if (IN_RELU) {
        a4.x = fmaxf(a4.x, 0.f); a4.y = fmaxf(a4.y, 0.f);
        a4.z = fmaxf(a4.z, 0.f); a4.w = fmaxf(a4.w, 0.f);
      }
      _Float16* dst = &As[arow * 32 + aseg];
      dst[0] = (_Float16)a4.x; dst[1] = (_Float16)a4.y;
      dst[2] = (_Float16)a4.z; dst[3] = (_Float16)a4.w;
    }
    // ---- stage B (32x64 f32 -> transposed 64x32 f16) ----
#pragma unroll
    for (int p = 0; p < 4; ++p) {
      const int kr = bkr + p * 8;     // 0..31
      const float* __restrict__ wr = W + (size_t)(k0 + kr) * N;
      float w0, w1, w2, w3;
      if (fullN) {
        float4 w4 = *(const float4*)(wr + nbase + bns);
        w0 = w4.x; w1 = w4.y; w2 = w4.z; w3 = w4.w;
        if (k0 + 32 < K)                      // -> global_prefetch_b8
          __builtin_prefetch(wr + (size_t)32 * N + nbase + bns, 0, 1);
      } else {
        const int c0 = nbase + bns;
        w0 = wr[(c0     < N) ? c0     : (N-1)];
        w1 = wr[(c0 + 1 < N) ? c0 + 1 : (N-1)];
        w2 = wr[(c0 + 2 < N) ? c0 + 2 : (N-1)];
        w3 = wr[(c0 + 3 < N) ? c0 + 3 : (N-1)];
      }
      Bs[(bns + 0) * 32 + kr] = (_Float16)w0;
      Bs[(bns + 1) * 32 + kr] = (_Float16)w1;
      Bs[(bns + 2) * 32 + kr] = (_Float16)w2;
      Bs[(bns + 3) * 32 + kr] = (_Float16)w3;
    }
    __syncthreads();
    // ---- fragments + WMMA ----
    if (active) {
      const v8h alo = *(const v8h*)&As[l16 * 32 + half * 8];
      const v8h ahi = *(const v8h*)&As[l16 * 32 + half * 8 + 16];
      const v8h blo = *(const v8h*)&Bs[(wv * 16 + l16) * 32 + half * 8];
      const v8h bhi = *(const v8h*)&Bs[(wv * 16 + l16) * 32 + half * 8 + 16];
      v16h af, bf;
#pragma unroll
      for (int i = 0; i < 8; ++i) {
        af[i] = alo[i]; af[8 + i] = ahi[i];
        bf[i] = blo[i]; bf[8 + i] = bhi[i];
      }
      acc = __builtin_amdgcn_wmma_f32_16x16x32_f16(false, af, false, bf,
                                                   (short)0, acc, false, false);
    }
    __syncthreads();
  }
  if (active) {
    const int bcol = ntile + l16;
#pragma unroll
    for (int r = 0; r < 8; ++r) {
      const int row = mtile * 16 + half * 8 + r;
      float v = acc[r] + bias[bcol];
      if (res) v += res[(size_t)row * N + bcol];
      if (OUT_RELU) v = fmaxf(v, 0.f);
      Y[(size_t)row * N + bcol] = v;
    }
  }
}

// ---- wave-per-row LayerNorm (optional fused residual) ---------------
__global__ void layernorm_k(const float* __restrict__ x,
                            const float* __restrict__ res,
                            const float* __restrict__ g,
                            const float* __restrict__ b,
                            float* __restrict__ y, int rows, int C)
{
  const int row  = blockIdx.x * (blockDim.x >> 5) + (threadIdx.x >> 5);
  const int lane = threadIdx.x & 31;
  if (row >= rows) return;
  const float* xr = x + (size_t)row * C;
  const float* rr = res ? res + (size_t)row * C : nullptr;
  float s = 0.f;
  for (int c = lane; c < C; c += 32) s += xr[c] + (rr ? rr[c] : 0.f);
  const float mean = wave_sum(s) / (float)C;
  float s2 = 0.f;
  for (int c = lane; c < C; c += 32) {
    float v = xr[c] + (rr ? rr[c] : 0.f) - mean; s2 += v * v;
  }
  const float inv = rsqrtf(wave_sum(s2) / (float)C + 1e-5f);
  float* yr = y + (size_t)row * C;
  for (int c = lane; c < C; c += 32) {
    float v = xr[c] + (rr ? rr[c] : 0.f);
    yr[c] = (v - mean) * inv * g[c] + b[c];
  }
}

// ---- point transforms (may run in place) ----------------------------
__global__ void pts_T44_k(const float* __restrict__ pl,
                          const float* __restrict__ T,   // [ntok,4,4]
                          float* __restrict__ out, int ntok, int npts)
{
  const int idx = blockIdx.x * blockDim.x + threadIdx.x;
  if (idx >= ntok * npts) return;
  const int tok = idx / npts;
  const float* Tt = T + (size_t)tok * 16;
  const float x = pl[(size_t)idx*3], y = pl[(size_t)idx*3+1], z = pl[(size_t)idx*3+2];
  out[(size_t)idx*3]   = Tt[0]*x + Tt[1]*y + Tt[2]*z  + Tt[3];
  out[(size_t)idx*3+1] = Tt[4]*x + Tt[5]*y + Tt[6]*z  + Tt[7];
  out[(size_t)idx*3+2] = Tt[8]*x + Tt[9]*y + Tt[10]*z + Tt[11];
}

__global__ void pts_quat_k(const float* __restrict__ pl,
                           const float* __restrict__ quat,
                           const float* __restrict__ trans,
                           float* __restrict__ out, int ntok, int npts)
{
  const int idx = blockIdx.x * blockDim.x + threadIdx.x;
  if (idx >= ntok * npts) return;
  const int tok = idx / npts;
  float R[9]; quat_rot(quat + (size_t)tok*4, R);
  const float* t = trans + (size_t)tok*3;
  const float x = pl[(size_t)idx*3], y = pl[(size_t)idx*3+1], z = pl[(size_t)idx*3+2];
  out[(size_t)idx*3]   = R[0]*x + R[1]*y + R[2]*z + t[0];
  out[(size_t)idx*3+1] = R[3]*x + R[4]*y + R[5]*z + t[1];
  out[(size_t)idx*3+2] = R[6]*x + R[7]*y + R[8]*z + t[2];
}

__global__ void init_rigids_k(float* __restrict__ quat,
                              float* __restrict__ trans, int ntok)
{
  const int t = blockIdx.x * blockDim.x + threadIdx.x;
  if (t >= ntok) return;
  quat[(size_t)t*4] = 1.f; quat[(size_t)t*4+1] = 0.f;
  quat[(size_t)t*4+2] = 0.f; quat[(size_t)t*4+3] = 0.f;
  trans[(size_t)t*3] = 0.f; trans[(size_t)t*3+1] = 0.f; trans[(size_t)t*3+2] = 0.f;
}

// ---- IPA attention: one wave per (b,h,q) ----------------------------
__global__ void ipa_attn_k(const float* __restrict__ q,     // [NTOK,192]
                           const float* __restrict__ qp,    // [NTOK,144] (global frame)
                           const float* __restrict__ kv,    // [NPTOK,384] (k|v per head)
                           const float* __restrict__ kvp,   // [NPTOK,432] (global frame)
                           const float* __restrict__ head_w,// [12]
                           const float* __restrict__ pmask, // [Bb,LP]
                           const float* __restrict__ quat,  // [NTOK,4]
                           const float* __restrict__ trans, // [NTOK,3]
                           float* __restrict__ feat,        // [NTOK,576]
                           float* __restrict__ att)         // [Bb,12,16,200]
{
  const int wid  = blockIdx.x * (blockDim.x >> 5) + (threadIdx.x >> 5);
  const int lane = threadIdx.x & 31;
  if (wid >= Bb * Hh * LQ) return;                 // wave-uniform
  const int qi = wid & (LQ - 1);
  const int h  = (wid >> 4) % Hh;
  const int bi = wid / (Hh * LQ);
  const int tok = bi * LQ + qi;

  float qv[CIPA];
  const float* qrow = q + (size_t)tok * (Hh*CIPA) + h * CIPA;
#pragma unroll
  for (int c = 0; c < CIPA; ++c) qv[c] = qrow[c];
  float qpv[PQ*3];
  const float* qpr = qp + (size_t)tok * (Hh*PQ*3) + h * PQ * 3;
#pragma unroll
  for (int c = 0; c < PQ*3; ++c) qpv[c] = qpr[c];

  float hw;                                        // softplus(head_w)*sqrt(1/36)
  { float x = head_w[h]; hw = (x > 20.f) ? x : log1pf(expf(x)); hw *= (1.f/6.f); }
  const float qscale = 0.17677669529663687f;       // sqrt(1/(2*CIPA))

  float sc[7];
#pragma unroll
  for (int j = 0; j < 7; ++j) {
    const int kk = lane + j * 32;
    if (kk < LP) {
      const size_t kt = (size_t)(bi * LP + kk);
      const float* kr = kv + kt * (2*Hh*CIPA) + h * (2*CIPA);
      float dot = 0.f;
#pragma unroll
      for (int c = 0; c < CIPA; ++c) dot += qv[c] * kr[c];
      const float* kpr = kvp + kt * (Hh*(PQ+PV)*3) + h * (PQ+PV) * 3;
      float d2 = 0.f;
#pragma unroll
      for (int p = 0; p < PQ; ++p) {
        float dx = qpv[p*3]   - kpr[p*3];
        float dy = qpv[p*3+1] - kpr[p*3+1];
        float dz = qpv[p*3+2] - kpr[p*3+2];
        d2 += dx*dx + dy*dy + dz*dz;
      }
      sc[j] = dot * qscale - 0.5f * hw * d2
              + (pmask[bi * LP + kk] - 1.f) * 100000.f;
    } else sc[j] = -INFINITY;
  }
  float m = -INFINITY;
#pragma unroll
  for (int j = 0; j < 7; ++j) m = fmaxf(m, sc[j]);
  m = wave_max(m);
  float den = 0.f;
#pragma unroll
  for (int j = 0; j < 7; ++j) { sc[j] = __expf(sc[j] - m); den += sc[j]; }
  den = wave_sum(den);
  const float rden = 1.f / den;

  float oacc[CIPA];  float opacc[PV*3];
#pragma unroll
  for (int c = 0; c < CIPA; ++c) oacc[c] = 0.f;
#pragma unroll
  for (int e = 0; e < PV*3; ++e) opacc[e] = 0.f;
#pragma unroll
  for (int j = 0; j < 7; ++j) {
    const int kk = lane + j * 32;
    if (kk < LP) {
      const float p = sc[j] * rden;
      att[(((size_t)bi * Hh + h) * LQ + qi) * LP + kk] = p;
      const size_t kt = (size_t)(bi * LP + kk);
      const float* vr  = kv  + kt * (2*Hh*CIPA) + h * (2*CIPA) + CIPA;
      const float* vpr = kvp + kt * (Hh*(PQ+PV)*3) + h * (PQ+PV) * 3 + PQ * 3;
#pragma unroll
      for (int c = 0; c < CIPA; ++c) oacc[c] += p * vr[c];
#pragma unroll
      for (int e = 0; e < PV*3; ++e) opacc[e] += p * vpr[e];
    }
  }
#pragma unroll
  for (int c = 0; c < CIPA; ++c) oacc[c] = wave_sum(oacc[c]);
#pragma unroll
  for (int e = 0; e < PV*3; ++e) opacc[e] = wave_sum(opacc[e]);

  float* fr = feat + (size_t)tok * FEATD;
  if (lane < CIPA) {
    fr[h * CIPA + lane] = oacc[lane];
  } else {
    const int pv = lane - 16;
    if (pv < PV) {
      float R[9]; quat_rot(quat + (size_t)tok*4, R);
      const float* t = trans + (size_t)tok*3;
      const float dx = opacc[pv*3]   - t[0];
      const float dy = opacc[pv*3+1] - t[1];
      const float dz = opacc[pv*3+2] - t[2];
      const float lx = R[0]*dx + R[3]*dy + R[6]*dz;   // R^T * d
      const float ly = R[1]*dx + R[4]*dy + R[7]*dz;
      const float lz = R[2]*dx + R[5]*dy + R[8]*dz;
      fr[Hh*CIPA + (h * PV + pv) * 3 + 0] = lx;
      fr[Hh*CIPA + (h * PV + pv) * 3 + 1] = ly;
      fr[Hh*CIPA + (h * PV + pv) * 3 + 2] = lz;
      fr[Hh*CIPA + Hh*PV*3 + h * PV + pv] = sqrtf(lx*lx + ly*ly + lz*lz + 1e-8f);
    }
  }
}

// ---- tiny dense heads (N=6 bb, N=14 angles) -------------------------
__global__ void small_gemm_k(const float* __restrict__ X,
                             const float* __restrict__ W,
                             const float* __restrict__ bias,
                             float* __restrict__ Y,
                             int M, int N, int K, int in_relu)
{
  const int idx = blockIdx.x * blockDim.x + threadIdx.x;
  if (idx >= M * N) return;
  const int r = idx / N, c = idx % N;
  const float* xr = X + (size_t)r * K;
  float s = bias[c];
  for (int k = 0; k < K; ++k) {
    float v = xr[k]; if (in_relu) v = fmaxf(v, 0.f);
    s += v * W[(size_t)k * N + c];
  }
  Y[idx] = s;
}

// ---- backbone rigid update + frames7 output -------------------------
__global__ void bb_update_k(const float* __restrict__ bbv,  // [NTOK,6]
                            float* __restrict__ quat, float* __restrict__ trans,
                            float* __restrict__ frames7, int ntok)
{
  const int t = blockIdx.x * blockDim.x + threadIdx.x;
  if (t >= ntok) return;
  const float* u = bbv + (size_t)t * 6;
  float q0 = quat[(size_t)t*4],   q1 = quat[(size_t)t*4+1];
  float q2 = quat[(size_t)t*4+2], q3 = quat[(size_t)t*4+3];
  float qtmp[4] = {q0, q1, q2, q3};
  float R[9]; quat_rot(qtmp, R);                  // R_old (pre-update)
  const float bx = u[0], by = u[1], bz = u[2];
  float nw = q0      - q1*bx - q2*by - q3*bz;
  float nx = q0*bx + q1      + q2*bz - q3*by;
  float ny = q0*by - q1*bz + q2      + q3*bx;
  float nz = q0*bz + q1*by - q2*bx + q3;
  const float inv = rsqrtf(nw*nw + nx*nx + ny*ny + nz*nz);
  nw *= inv; nx *= inv; ny *= inv; nz *= inv;
  const float tx = trans[(size_t)t*3]   + R[0]*u[3] + R[1]*u[4] + R[2]*u[5];
  const float ty = trans[(size_t)t*3+1] + R[3]*u[3] + R[4]*u[4] + R[5]*u[5];
  const float tz = trans[(size_t)t*3+2] + R[6]*u[3] + R[7]*u[4] + R[8]*u[5];
  quat[(size_t)t*4]=nw; quat[(size_t)t*4+1]=nx; quat[(size_t)t*4+2]=ny; quat[(size_t)t*4+3]=nz;
  trans[(size_t)t*3]=tx; trans[(size_t)t*3+1]=ty; trans[(size_t)t*3+2]=tz;
  float* f7 = frames7 + (size_t)t * 7;
  f7[0]=nw; f7[1]=nx; f7[2]=ny; f7[3]=nz;
  f7[4]=tx*10.f; f7[5]=ty*10.f; f7[6]=tz*10.f;
}

// ---- angle normalization (unnorm -> ang) ----------------------------
__global__ void ang_norm_k(const float* __restrict__ u,
                           float* __restrict__ ang, int n)
{
  const int i = blockIdx.x * blockDim.x + threadIdx.x;
  if (i >= n) return;
  const float a = u[2*i], b = u[2*i+1];
  const float d = sqrtf(fmaxf(a*a + b*b, 1e-8f));
  ang[2*i] = a / d; ang[2*i+1] = b / d;
}

// ---- torsion frames + sc44 + atom positions -------------------------
__device__ __forceinline__ void frame_comp(const float* Ra, const float* ta,
                                           const float* Rb, const float* tb,
                                           float* Ro, float* to)
{
#pragma unroll
  for (int r = 0; r < 3; ++r) {
#pragma unroll
    for (int c = 0; c < 3; ++c)
      Ro[r*3+c] = Ra[r*3]*Rb[c] + Ra[r*3+1]*Rb[3+c] + Ra[r*3+2]*Rb[6+c];
    to[r] = Ra[r*3]*tb[0] + Ra[r*3+1]*tb[1] + Ra[r*3+2]*tb[2] + ta[r];
  }
}

__global__ void torsion_atoms_k(const int*   __restrict__ aatype,
                                const float* __restrict__ quat,
                                const float* __restrict__ trans,
                                const float* __restrict__ ang,       // [NTOK,7,2]
                                const float* __restrict__ def,       // [21,8,4,4]
                                const int*   __restrict__ gidx,      // [21,18]
                                const float* __restrict__ amask,     // [21,18]
                                const float* __restrict__ lit,       // [21,18,3]
                                float* __restrict__ sc44,            // [NTOK,8,4,4]
                                float* __restrict__ xyz, int ntok)   // [NTOK,18,3]
{
  const int t = blockIdx.x * blockDim.x + threadIdx.x;
  if (t >= ntok) return;
  const int aa = aatype[t];
  float Rbb[9]; quat_rot(quat + (size_t)t*4, Rbb);
  const float tb0 = trans[(size_t)t*3]*10.f, tb1 = trans[(size_t)t*3+1]*10.f,
              tb2 = trans[(size_t)t*3+2]*10.f;

  float Rl[NGRP][9], tl[NGRP][3];
#pragma unroll
  for (int g = 0; g < NGRP; ++g) {
    const float* df = def + ((size_t)aa * NGRP + g) * 16;
    float s_ = 0.f, c_ = 1.f;
    if (g > 0) { s_ = ang[(size_t)t*14 + (g-1)*2]; c_ = ang[(size_t)t*14 + (g-1)*2 + 1]; }
#pragma unroll
    for (int r = 0; r < 3; ++r) {
      const float d0 = df[r*4], d1 = df[r*4+1], d2 = df[r*4+2];
      Rl[g][r*3+0] = d0;
      Rl[g][r*3+1] = d1 * c_ + d2 * s_;
      Rl[g][r*3+2] = -d1 * s_ + d2 * c_;
      tl[g][r] = df[r*4+3];
    }
  }
  float Rc[NGRP][9], tc[NGRP][3];
#pragma unroll
  for (int g = 0; g < 5; ++g) {
#pragma unroll
    for (int e = 0; e < 9; ++e) Rc[g][e] = Rl[g][e];
#pragma unroll
    for (int e = 0; e < 3; ++e) tc[g][e] = tl[g][e];
  }
  frame_comp(Rc[4], tc[4], Rl[5], tl[5], Rc[5], tc[5]);
  frame_comp(Rc[5], tc[5], Rl[6], tl[6], Rc[6], tc[6]);
  frame_comp(Rc[6], tc[6], Rl[7], tl[7], Rc[7], tc[7]);

  float Rg[NGRP][9], tg[NGRP][3];
#pragma unroll
  for (int g = 0; g < NGRP; ++g) {
#pragma unroll
    for (int r = 0; r < 3; ++r) {
#pragma unroll
      for (int c = 0; c < 3; ++c)
        Rg[g][r*3+c] = Rbb[r*3]*Rc[g][c] + Rbb[r*3+1]*Rc[g][3+c] + Rbb[r*3+2]*Rc[g][6+c];
    }
    tg[g][0] = Rbb[0]*tc[g][0] + Rbb[1]*tc[g][1] + Rbb[2]*tc[g][2] + tb0;
    tg[g][1] = Rbb[3]*tc[g][0] + Rbb[4]*tc[g][1] + Rbb[5]*tc[g][2] + tb1;
    tg[g][2] = Rbb[6]*tc[g][0] + Rbb[7]*tc[g][1] + Rbb[8]*tc[g][2] + tb2;
    float* s = sc44 + ((size_t)t * NGRP + g) * 16;
#pragma unroll
    for (int r = 0; r < 3; ++r) {
      s[r*4]=Rg[g][r*3]; s[r*4+1]=Rg[g][r*3+1]; s[r*4+2]=Rg[g][r*3+2]; s[r*4+3]=tg[g][r];
    }
    s[12]=0.f; s[13]=0.f; s[14]=0.f; s[15]=1.f;
  }
#pragma unroll
  for (int a = 0; a < NAT; ++a) {
    const int g = gidx[aa * NAT + a];
    const float* lp = lit + ((size_t)aa * NAT + a) * 3;
    const float m = amask[aa * NAT + a];
    const float px = lp[0], py = lp[1], pz = lp[2];
    float* o = xyz + ((size_t)t * NAT + a) * 3;
    o[0] = (Rg[g][0]*px + Rg[g][1]*py + Rg[g][2]*pz + tg[g][0]) * m;
    o[1] = (Rg[g][3]*px + Rg[g][4]*py + Rg[g][5]*pz + tg[g][1]) * m;
    o[2] = (Rg[g][6]*px + Rg[g][7]*py + Rg[g][8]*pz + tg[g][2]) * m;
  }
}

// =====================================================================
// Host orchestration
// =====================================================================
template<int K>
static void launch_gemmK(hipStream_t st, const float* X, const float* W,
                         const float* b, const float* res, float* Y,
                         int M, int N, int ir, int orl)
{
  dim3 grid((N + 63) / 64, M / 16), blk(128);
  if (!ir && !orl)      gemm_tile_k<K,false,false><<<grid, blk, 0, st>>>(X, W, b, res, Y, M, N);
  else if (!ir && orl)  gemm_tile_k<K,false,true ><<<grid, blk, 0, st>>>(X, W, b, res, Y, M, N);
  else if (ir && !orl)  gemm_tile_k<K,true ,false><<<grid, blk, 0, st>>>(X, W, b, res, Y, M, N);
  else                  gemm_tile_k<K,true ,true ><<<grid, blk, 0, st>>>(X, W, b, res, Y, M, N);
}
static inline void launch_ln(hipStream_t st, const float* x, const float* res,
                             const float* g, const float* b, float* y, int rows, int C)
{
  layernorm_k<<<dim3((rows + 7) / 8), dim3(256), 0, st>>>(x, res, g, b, y, rows, C);
}

extern "C" void kernel_launch(void* const* d_in, const int* in_sizes, int n_in,
                              void* d_out, int out_size, void* d_ws, size_t ws_size,
                              hipStream_t stream)
{
  (void)in_sizes; (void)n_in; (void)out_size; (void)ws_size;
  #define FI(i) ((const float*)d_in[(i)])
  #define II(i) ((const int*)d_in[(i)])

  float* ws = (float*)d_ws;
  size_t off = 0;
  auto alloc = [&](size_t n) { float* p = ws + off; off += n; return p; };

  float* s_init  = alloc((size_t)NTOK  * CS);
  float* sA      = alloc((size_t)NTOK  * CS);   // running s_loop
  float* sB      = alloc((size_t)NTOK  * CS);   // post-IPA LN scratch
  float* prot_ln = alloc((size_t)NPTOK * CS);   // reused as kv buffer
  float* s_prot  = alloc((size_t)NPTOK * CS);
  float* kvp     = alloc((size_t)NPTOK * (Hh*(PQ+PV)*3)); // 432
  float* qb      = alloc((size_t)NTOK  * (Hh*CIPA));       // 192
  float* qp      = alloc((size_t)NTOK  * (Hh*PQ*3));       // 144
  float* feat    = alloc((size_t)NTOK  * FEATD);           // 576
  float* upd     = alloc((size_t)NTOK  * CS);
  float* t1      = alloc((size_t)NTOK  * CS);
  float* t2      = alloc((size_t)NTOK  * CS);
  float* quat    = alloc((size_t)NTOK  * 4);
  float* trans   = alloc((size_t)NTOK  * 3);
  float* bbv     = alloc((size_t)NTOK  * 6);
  float* angA    = alloc((size_t)NTOK  * CRES);
  float* angB    = alloc((size_t)NTOK  * CRES);
  float* kvb     = prot_ln;   // prot_ln is dead after s_prot projection

  float* out         = (float*)d_out;
  float* out_sloop   = out;                 // 1024*384
  float* out_frames7 = out + 393216;        // 1024*7
  float* out_sc44    = out + 400384;        // 1024*8*16
  float* out_unnorm  = out + 531456;        // 1024*14
  float* out_ang     = out + 545792;        // 1024*14
  float* out_xyz     = out + 560128;        // 1024*18*3
  float* out_att     = out + 615424;        // 64*12*16*200

  // ---- loop-invariant prologue ----
  launch_ln(stream, FI(IN_S_LOOP_INIT), nullptr, FI(IN_LNLOOP_G), FI(IN_LNLOOP_B),
            s_init, NTOK, CS);
  launch_ln(stream, FI(IN_S_PROT_INIT), nullptr, FI(IN_LNPROT_G), FI(IN_LNPROT_B),
            prot_ln, NPTOK, CS);
  launch_gemmK<CS>(stream, s_init,  FI(IN_INLOOP_W), FI(IN_INLOOP_B), nullptr, sA,
                   NTOK, CS, 0, 0);
  launch_gemmK<CS>(stream, prot_ln, FI(IN_INPROT_W), FI(IN_INPROT_B), nullptr, s_prot,
                   NPTOK, CS, 0, 0);
  // hoisted: k/v and k/v points do not change across blocks
  launch_gemmK<CS>(stream, s_prot, FI(IN_KVP_W), FI(IN_KVP_B), nullptr, kvp,
                   NPTOK, Hh*(PQ+PV)*3, 0, 0);
  launch_gemmK<CS>(stream, s_prot, FI(IN_KV_W), FI(IN_KV_B), nullptr, kvb,
                   NPTOK, 2*Hh*CIPA, 0, 0);
  {
    const int n = NPTOK * Hh * (PQ+PV);
    pts_T44_k<<<dim3((n + 255) / 256), dim3(256), 0, stream>>>(
        kvp, FI(IN_T_PROTEIN), kvp, NPTOK, Hh*(PQ+PV));
  }
  init_rigids_k<<<dim3(4), dim3(256), 0, stream>>>(quat, trans, NTOK);

  // ---- NBLOCKS iterations ----
  for (int blk = 0; blk < 2; ++blk) {
    launch_gemmK<CS>(stream, sA, FI(IN_Q_W),  FI(IN_Q_B),  nullptr, qb, NTOK, Hh*CIPA, 0, 0);
    launch_gemmK<CS>(stream, sA, FI(IN_QP_W), FI(IN_QP_B), nullptr, qp, NTOK, Hh*PQ*3, 0, 0);
    {
      const int n = NTOK * Hh * PQ;
      pts_quat_k<<<dim3((n + 255) / 256), dim3(256), 0, stream>>>(
          qp, quat, trans, qp, NTOK, Hh*PQ);
    }
    ipa_attn_k<<<dim3(Bb*Hh*LQ/4), dim3(128), 0, stream>>>(
        qb, qp, kvb, kvp, FI(IN_HEAD_W), FI(IN_PROT_MASK), quat, trans,
        feat, out_att);
    launch_gemmK<FEATD>(stream, feat, FI(IN_IPAOUT_W), FI(IN_IPAOUT_B), nullptr, upd,
                        NTOK, CS, 0, 0);
    launch_ln(stream, sA, upd, FI(IN_LNIPA_G), FI(IN_LNIPA_B), sB, NTOK, CS);
    // transition MLP (NTRANS=1)
    launch_gemmK<CS>(stream, sB, FI(IN_TL_L1_W), FI(IN_TL_L1_B), nullptr, t1, NTOK, CS, 0, 1);
    launch_gemmK<CS>(stream, t1, FI(IN_TL_L2_W), FI(IN_TL_L2_B), nullptr, t2, NTOK, CS, 0, 1);
    launch_gemmK<CS>(stream, t2, FI(IN_TL_L3_W), FI(IN_TL_L3_B), sB,      sA, NTOK, CS, 0, 0);
    launch_ln(stream, sA, nullptr, FI(IN_LNTRANS_G), FI(IN_LNTRANS_B), sA, NTOK, CS);
    // backbone update
    small_gemm_k<<<dim3((NTOK*6 + 255) / 256), dim3(256), 0, stream>>>(
        sA, FI(IN_BB_W), FI(IN_BB_B), bbv, NTOK, 6, CS, 0);
    bb_update_k<<<dim3(4), dim3(256), 0, stream>>>(bbv, quat, trans, out_frames7, NTOK);
    // angle resnet
    launch_gemmK<CS>(stream, sA,     FI(IN_ANG_IN_W),   FI(IN_ANG_IN_B),   nullptr, angA,
                     NTOK, CRES, 1, 0);
    launch_gemmK<CS>(stream, s_init, FI(IN_ANG_INIT_W), FI(IN_ANG_INIT_B), angA,    angA,
                     NTOK, CRES, 1, 0);
    launch_gemmK<CRES>(stream, angA, FI(IN_ANG_B0_L1_W), FI(IN_ANG_B0_L1_B), nullptr, angB,
                       NTOK, CRES, 1, 0);
    launch_gemmK<CRES>(stream, angB, FI(IN_ANG_B0_L2_W), FI(IN_ANG_B0_L2_B), angA,    angA,
                       NTOK, CRES, 1, 0);
    launch_gemmK<CRES>(stream, angA, FI(IN_ANG_B1_L1_W), FI(IN_ANG_B1_L1_B), nullptr, angB,
                       NTOK, CRES, 1, 0);
    launch_gemmK<CRES>(stream, angB, FI(IN_ANG_B1_L2_W), FI(IN_ANG_B1_L2_B), angA,    angA,
                       NTOK, CRES, 1, 0);
    small_gemm_k<<<dim3((NTOK*14 + 255) / 256), dim3(256), 0, stream>>>(
        angA, FI(IN_ANG_OUT_W), FI(IN_ANG_OUT_B), out_unnorm, NTOK, 2*NANG, CRES, 1);
    ang_norm_k<<<dim3((NTOK*NANG + 255) / 256), dim3(256), 0, stream>>>(
        out_unnorm, out_ang, NTOK * NANG);
    torsion_atoms_k<<<dim3(8), dim3(128), 0, stream>>>(
        II(IN_AATYPE), quat, trans, out_ang, FI(IN_DEF_FRAMES), II(IN_GROUP_IDX),
        FI(IN_ATOM_MASK), FI(IN_LIT_POS), out_sc44, out_xyz, NTOK);
  }
  hipMemcpyAsync(out_sloop, sA, (size_t)NTOK * CS * sizeof(float),
                 hipMemcpyDeviceToDevice, stream);
  #undef FI
  #undef II
}
